// RevGATBlock_86517821214617
// MI455X (gfx1250) — matrix operand order repaired and checked by reading, not compiled, not run on hardware
//
#include <hip/hip_runtime.h>
#include <stdint.h>

#define CC 256
#define HH 4
#define DD 64
#define EPSV 1e-5f
#define SLOPEV 0.2f
#define ENC_NEG_INF 0x007FFFFFu

typedef float    v8f  __attribute__((ext_vector_type(8)));
typedef _Float16 v16h __attribute__((ext_vector_type(16)));
typedef _Float16 v8h  __attribute__((ext_vector_type(8)));

__device__ __forceinline__ unsigned short f2h_bits(float f) {
  _Float16 h = (_Float16)f;
  unsigned short s;
  __builtin_memcpy(&s, &h, 2);
  return s;
}
__device__ __forceinline__ float h2f(unsigned short s) {
  _Float16 h;
  __builtin_memcpy(&h, &s, 2);
  return (float)h;
}
// order-preserving f32 -> u32 encoding (for atomicMax on signed floats)
__device__ __forceinline__ unsigned fenc(float f) {
  unsigned u = __float_as_uint(f);
  return (u & 0x80000000u) ? ~u : (u | 0x80000000u);
}
__device__ __forceinline__ float fdec(unsigned u) {
  unsigned v = (u & 0x80000000u) ? (u & 0x7FFFFFFFu) : ~u;
  return __uint_as_float(v);
}

// ---------------- 1) BN + ReLU + dropout mask -> h (f16) ----------------
__global__ void bn_relu_f16(const float* __restrict__ x,
                            const float* __restrict__ gamma,
                            const float* __restrict__ beta,
                            const float* __restrict__ mean,
                            const float* __restrict__ var,
                            const float* __restrict__ mask,
                            unsigned short* __restrict__ hbf, int total) {
  int i = blockIdx.x * 256 + threadIdx.x;
  if (i >= total) return;
  int c = i & (CC - 1);
  float sc = gamma[c] * rsqrtf(var[c] + EPSV);
  float v = (x[i] - mean[c]) * sc + beta[c];
  v = fmaxf(v, 0.f) * mask[i];
  hbf[i] = f2h_bits(v);
}

// ------- 2) pack W2 = [fc_w ; fc_dst_w] into WMMA B-fragment order -------
// B frag (32x16 KxN): lane<16 -> N=lane, elems j: K=kt*32+(j&7)+(j>>3)*16 ; lane>=16 -> K+8
__global__ void pack_weights(const float* __restrict__ fc_w,
                             const float* __restrict__ fc_dst_w,
                             unsigned short* __restrict__ packedB) {
  int idx = blockIdx.x * 256 + threadIdx.x;            // 32*8*32*16 = 131072
  if (idx >= 32 * 8 * 32 * 16) return;
  int j     = idx & 15;
  int lane  = (idx >> 4) & 31;
  int kt    = (idx >> 9) & 7;
  int ntile = idx >> 12;                               // 0..31 (512 cols / 16)
  int col = ntile * 16 + (lane & 15);                  // 0..511
  int k   = kt * 32 + (lane >> 4) * 8 + (j & 7) + (j >> 3) * 16; // 0..255
  const float* W = (col < CC) ? fc_w : fc_dst_w;
  int c = col & (CC - 1);
  packedB[idx] = f2h_bits(W[c * CC + k]);
}

// ---------------- 3) init: zero d_out, denom; emax = enc(-inf) ----------------
__global__ void init_bufs(float* __restrict__ out, float* __restrict__ denom,
                          unsigned* __restrict__ emax, int totalNC, int totalNH) {
  int i = blockIdx.x * 256 + threadIdx.x;
  if (i < totalNC) out[i] = 0.f;
  if (i < totalNH) { denom[i] = 0.f; emax[i] = ENC_NEG_INF; }
}

// ---------------- 4) fused GEMM: [N,256]x[256,512] via WMMA f16 ----------------
// block = 256 thr = 8 waves; wave w covers cols [w*64, w*64+64):
//   w<4: feat_src head w (store f16 tile + el partial dot with attn_l)
//   w>=4: feat_dst head w-4 (no store; only er partial dot with attn_r)
__global__ void gat_gemm_wmma(const unsigned short* __restrict__ hbf,
                              const unsigned short* __restrict__ packedB,
                              const float* __restrict__ attn_l,
                              const float* __restrict__ attn_r,
                              unsigned short* __restrict__ feat_src,
                              float* __restrict__ el, float* __restrict__ er,
                              int N) {
  __shared__ unsigned lds32[2048];                     // 16 rows x 256 f16 = 8KB
  const int mtile = blockIdx.x;
  const int mbase = mtile * 16;
  const int tid = threadIdx.x;
  const int wave = tid >> 5, lane = tid & 31;

  // cooperative stage of A strip into LDS (zero-fill past N)
  const unsigned* src32 = (const unsigned*)(hbf + (size_t)mbase * CC);
  for (int i = tid; i < 2048; i += 256) {
    int row = i >> 7;                                   // 128 dwords per row
    lds32[i] = (mbase + row < N) ? src32[i] : 0u;
  }
  __syncthreads();

  const bool is_src = (wave < 4);
  const int  h      = wave & 3;
  const int  krow   = lane >> 4;                        // 0/1 -> K-half of A frag
  const int  arow   = lane & 15;
  const _Float16* ldsA = (const _Float16*)lds32;

  v8f acc[4] = {};
  #pragma unroll
  for (int kt = 0; kt < 8; ++kt) {
    const int koff = kt * 32 + krow * 8;
    v8h lo = *(const v8h*)(ldsA + arow * CC + koff);
    v8h hi = *(const v8h*)(ldsA + arow * CC + koff + 16);
    v16h a = __builtin_shufflevector(lo, hi, 0, 1, 2, 3, 4, 5, 6, 7,
                                             8, 9, 10, 11, 12, 13, 14, 15);
    #pragma unroll
    for (int t = 0; t < 4; ++t) {
      const int ntile = wave * 4 + t;                   // 0..31
      v16h b = *(const v16h*)((const _Float16*)packedB +
                              ((size_t)(ntile * 8 + kt) * 32 + lane) * 16);
      acc[t] = __builtin_amdgcn_wmma_f32_16x16x32_f16(
          false, a, false, b, (short)0, acc[t], false, false);
    }
  }

  // epilogue: per-row dot with attn vector (64 cols of this head)
  const float* attn = is_src ? attn_l : attn_r;         // [H][64]
  const int cl = lane & 15;
  float aw[4];
  #pragma unroll
  for (int t = 0; t < 4; ++t) aw[t] = attn[h * DD + t * 16 + cl];

  float part[8];
  #pragma unroll
  for (int j = 0; j < 8; ++j) {
    float p = 0.f;
    #pragma unroll
    for (int t = 0; t < 4; ++t) p += acc[t][j] * aw[t];
    // reduce across the 16 lanes of this row-half (xor<=8 stays in half)
    p += __shfl_xor(p, 1, 32);
    p += __shfl_xor(p, 2, 32);
    p += __shfl_xor(p, 4, 32);
    p += __shfl_xor(p, 8, 32);
    part[j] = p;
  }
  float* elp = is_src ? el : er;
  if (cl == 0) {
    const int rowbase = mbase + krow * 8;
    #pragma unroll
    for (int j = 0; j < 8; ++j)
      if (rowbase + j < N) elp[(size_t)(rowbase + j) * HH + h] = part[j];
  }
  if (is_src) {
    #pragma unroll
    for (int j = 0; j < 8; ++j) {
      const int row = mbase + krow * 8 + j;
      if (row < N) {
        #pragma unroll
        for (int t = 0; t < 4; ++t)
          feat_src[(size_t)row * CC + h * DD + t * 16 + cl] = f2h_bits(acc[t][j]);
      }
    }
  }
}

// ---------------- 5) edge pass A: segment max via encoded atomicMax ----------------
__global__ void edge_max_k(const int* __restrict__ ei,
                           const float* __restrict__ el,
                           const float* __restrict__ er,
                           unsigned* __restrict__ emax, int E) {
  int idx = blockIdx.x * 256 + threadIdx.x;
  if (idx >= E * HH) return;
  int e = idx >> 2, h = idx & 3;
  int s = ei[e], d = ei[E + e];
  float v = el[(size_t)s * HH + h] + er[(size_t)d * HH + h];
  v = (v > 0.f) ? v : SLOPEV * v;
  atomicMax(&emax[(size_t)d * HH + h], fenc(v));
}

// ------- 6) edge pass B: w=exp(e-max); denom += w; out[dst] += w*feat_src[src] -------
// one wave per edge (256 feats = 8 coalesced iters of 32 lanes)
__global__ void edge_scatter_k(const int* __restrict__ ei,
                               const float* __restrict__ el,
                               const float* __restrict__ er,
                               const unsigned* __restrict__ emax,
                               const unsigned short* __restrict__ feat_src,
                               float* __restrict__ denom,
                               float* __restrict__ out, int E) {
  int gid = blockIdx.x * 256 + threadIdx.x;
  int e = gid >> 5, lane = gid & 31;
  if (e >= E) return;
  int s = ei[e], d = ei[E + e];

  float wv = 0.f;
  if (lane < HH) {
    float v = el[(size_t)s * HH + lane] + er[(size_t)d * HH + lane];
    v = (v > 0.f) ? v : SLOPEV * v;
    float m = fdec(emax[(size_t)d * HH + lane]);
    wv = __expf(v - m);
    atomicAdd(&denom[(size_t)d * HH + lane], wv);
  }
  float wc[HH];
  #pragma unroll
  for (int hh = 0; hh < HH; ++hh) wc[hh] = __shfl(wv, hh, 32);

  const unsigned short* fs = feat_src + (size_t)s * CC;
  float* o = out + (size_t)d * CC;
  #pragma unroll
  for (int it = 0; it < 8; ++it) {
    int c = it * 32 + lane;
    atomicAdd(&o[c], wc[c >> 6] * h2f(fs[c]));
  }
}

// ---------------- 7) finalize: out = agg/denom + BN(x) + bias ----------------
__global__ void finalize_k(const float* __restrict__ x,
                           const float* __restrict__ gamma,
                           const float* __restrict__ beta,
                           const float* __restrict__ mean,
                           const float* __restrict__ var,
                           const float* __restrict__ mask,
                           const float* __restrict__ bias,
                           const float* __restrict__ denom,
                           float* __restrict__ out, int total) {
  int i = blockIdx.x * 256 + threadIdx.x;
  if (i >= total) return;
  int c = i & (CC - 1);
  int n = i >> 8;
  int h = c >> 6;
  float dn = denom[(size_t)n * HH + h];
  float r = (dn > 0.f) ? out[i] / dn : 0.f;
  float sc = gamma[c] * rsqrtf(var[c] + EPSV);
  float hv = (x[i] - mean[c]) * sc + beta[c];
  hv = fmaxf(hv, 0.f) * mask[i];
  out[i] = r + hv + bias[c];
}

extern "C" void kernel_launch(void* const* d_in, const int* in_sizes, int n_in,
                              void* d_out, int out_size, void* d_ws, size_t ws_size,
                              hipStream_t stream) {
  const float* x        = (const float*)d_in[0];
  const int*   ei       = (const int*)d_in[1];
  const float* mask     = (const float*)d_in[2];
  const float* gamma    = (const float*)d_in[3];
  const float* beta     = (const float*)d_in[4];
  const float* mean     = (const float*)d_in[5];
  const float* var      = (const float*)d_in[6];
  const float* fc_w     = (const float*)d_in[7];
  const float* fc_dst_w = (const float*)d_in[8];
  const float* attn_l   = (const float*)d_in[9];
  const float* attn_r   = (const float*)d_in[10];
  const float* bias     = (const float*)d_in[11];
  float* out = (float*)d_out;

  const int N = in_sizes[0] / CC;
  const int E = in_sizes[1] / 2;

  char* ws = (char*)d_ws;
  size_t off = 0;
  auto take = [&](size_t bytes) -> void* {
    void* p = ws + off;
    off += (bytes + 255) & ~(size_t)255;
    return p;
  };
  unsigned short* hbf     = (unsigned short*)take((size_t)N * CC * 2);
  unsigned short* featsrc = (unsigned short*)take((size_t)N * CC * 2);
  unsigned short* packedB = (unsigned short*)take((size_t)32 * 8 * 32 * 16 * 2);
  float*    el    = (float*)take((size_t)N * HH * 4);
  float*    er    = (float*)take((size_t)N * HH * 4);
  unsigned* emax  = (unsigned*)take((size_t)N * HH * 4);
  float*    denom = (float*)take((size_t)N * HH * 4);

  const int totalNC = N * CC;
  const int totalNH = N * HH;

  pack_weights<<<(32 * 8 * 32 * 16 + 255) / 256, 256, 0, stream>>>(
      fc_w, fc_dst_w, packedB);
  bn_relu_f16<<<(totalNC + 255) / 256, 256, 0, stream>>>(
      x, gamma, beta, mean, var, mask, hbf, totalNC);
  init_bufs<<<(totalNC + 255) / 256, 256, 0, stream>>>(
      out, denom, emax, totalNC, totalNH);
  gat_gemm_wmma<<<(N + 15) / 16, 256, 0, stream>>>(
      hbf, packedB, attn_l, attn_r, featsrc, el, er, N);
  edge_max_k<<<(E * HH + 255) / 256, 256, 0, stream>>>(ei, el, er, emax, E);
  edge_scatter_k<<<((size_t)E * 32 + 255) / 256, 256, 0, stream>>>(
      ei, el, er, emax, featsrc, denom, out, E);
  finalize_k<<<(totalNC + 255) / 256, 256, 0, stream>>>(
      x, gamma, beta, mean, var, mask, bias, denom, out, totalNC);
}
